// _CausalSelfAttentionRelativePosition_20650202759558
// MI455X (gfx1250) — matrix operand order repaired and checked by reading, not compiled, not run on hardware
//
#include <hip/hip_runtime.h>
#include <hip/hip_bf16.h>

#define N_HEAD 16
#define L_SEQ  2048
#define DMODEL 1024
#define HS     64
#define BATCH  2

typedef __attribute__((ext_vector_type(16))) __bf16 v16bf;
typedef __attribute__((ext_vector_type(8)))  __bf16 v8bf;
typedef __attribute__((ext_vector_type(8)))  float  v8f;
typedef __attribute__((ext_vector_type(4)))  unsigned int u32x4;
typedef __attribute__((ext_vector_type(8)))  int i32x8;
typedef __attribute__((ext_vector_type(4)))  int i32x4;

union BF16Frag { v16bf v; v8bf h[2]; };

__device__ __forceinline__ v16bf make_frag(v8bf lo, v8bf hi) {
    BF16Frag f; f.h[0] = lo; f.h[1] = hi; return f.v;
}

__device__ __forceinline__ v8f wmma_bf16(v16bf a, v16bf b, v8f c) {
    return __builtin_amdgcn_wmma_f32_16x16x32_bf16(false, a, false, b, (short)0, c, false, false);
}

// Tensor Data Mover: 2-D tile (tile_d0 fastest dim) from global -> LDS.
// D# packing per CDNA5 ISA 8.3/8.4 (group0: count/lds/global/type, group1:
// data_size=2B, dims, tile dims, dim0 stride). Groups 2/3 zero (<=2D tensor).
__device__ __forceinline__ void tdm_load_2d(unsigned lds_addr, const void* gptr,
                                            unsigned tensor_d0, unsigned tensor_d1,
                                            unsigned stride0,
                                            unsigned tile_d0, unsigned tile_d1) {
    unsigned long long ga = (unsigned long long)gptr;
    u32x4 g0;
    g0[0] = 1u;                                           // count=1, user mode
    g0[1] = lds_addr;                                     // LDS byte address
    g0[2] = (unsigned)(ga & 0xFFFFFFFFu);                 // global addr [31:0]
    g0[3] = (unsigned)((ga >> 32) & 0x01FFFFFFu) | (2u << 30); // [56:32] | type=2
    i32x8 g1;
    g1[0] = (int)(1u << 16);                              // data_size = 1 -> 2 bytes
    g1[1] = (int)((tensor_d0 & 0xFFFFu) << 16);           // tensor_dim0 [15:0]
    g1[2] = (int)(((tensor_d0 >> 16) & 0xFFFFu) | ((tensor_d1 & 0xFFFFu) << 16));
    g1[3] = (int)(((tensor_d1 >> 16) & 0xFFFFu) | ((tile_d0 & 0xFFFFu) << 16));
    g1[4] = (int)(tile_d1 & 0xFFFFu);                     // tile_dim1 (tile_dim2=0)
    g1[5] = (int)stride0;                                 // tensor_dim0_stride [31:0]
    g1[6] = 0;                                            // stride0 hi / stride1 lo
    g1[7] = 0;
    i32x4 z4 = {0, 0, 0, 0};
    i32x8 z8 = {0, 0, 0, 0, 0, 0, 0, 0};
    __builtin_amdgcn_tensor_load_to_lds(g0, g1, z4, z4, z8, 0);
}

// ---------------------------------------------------------------- converts
__global__ void cvt_bf16_kernel(const float* __restrict__ s, __bf16* __restrict__ d, int n) {
    int i = blockIdx.x * blockDim.x + threadIdx.x;
    int stride = gridDim.x * blockDim.x;
    for (; i < n; i += stride) d[i] = (__bf16)s[i];
}

// s: [R][C] f32 row-major -> d: [C][R] bf16 (write-coalesced)
__global__ void cvt_bf16_T_kernel(const float* __restrict__ s, __bf16* __restrict__ d,
                                  int R, int C) {
    int i = blockIdx.x * blockDim.x + threadIdx.x;
    const int total = R * C;
    const int stride = gridDim.x * blockDim.x;
    for (; i < total; i += stride) {
        const int c = i / R, r = i % R;
        d[i] = (__bf16)s[(size_t)r * C + c];
    }
}

// ---------------------------------------------------------------- QKV GEMM
// A [M=B*L, K=D] bf16 ; WT [N=3D, K=D] bf16 (K-major!) ; bias f32
// 4 waves/block, block tile 16 x 256, K slab of 32 staged to LDS by TDM.
__global__ void qkv_gemm_kernel(const __bf16* __restrict__ A,
                                const __bf16* __restrict__ WT,
                                const float*  __restrict__ bias,
                                __bf16* __restrict__ qo,
                                __bf16* __restrict__ ko,
                                __bf16* __restrict__ vt) {
    __shared__ __attribute__((aligned(16))) __bf16 sWT[2][256][32];   // 32 KB

    const int Kd = DMODEL;
    const int lane = threadIdx.x & 31;
    const int wave = threadIdx.x >> 5;
    const int m0 = blockIdx.x * 16;
    const int n0blk = blockIdx.y * 256;
    const int lid  = lane & 15;
    const int hi16 = lane >> 4;
    const int row  = m0 + lid;
    const int ak0  = hi16 * 8;
    const int kfrag = hi16 * 8;

    v8f acc[4];
    for (int t = 0; t < 4; ++t) acc[t] = {};

    if (wave == 0)
        tdm_load_2d((unsigned)(size_t)&sWT[0][0][0],
                    WT + (size_t)n0blk * Kd, Kd, 3 * DMODEL, Kd, 32, 256);

    for (int kb = 0; kb < Kd; kb += 32) {
        const int cur = (kb >> 5) & 1;
        const bool more = (kb + 32) < Kd;
        if (wave == 0) {
            if (more) {
                tdm_load_2d((unsigned)(size_t)&sWT[cur ^ 1][0][0],
                            WT + (size_t)n0blk * Kd + (kb + 32), Kd, 3 * DMODEL, Kd, 32, 256);
                __builtin_amdgcn_s_wait_tensorcnt(1);
            } else {
                __builtin_amdgcn_s_wait_tensorcnt(0);
            }
        }
        __syncthreads();

        if (kb + 64 < Kd) __builtin_prefetch(A + (size_t)row * Kd + kb + 64, 0, 1);
        v8bf alo = *(const v8bf*)(A + (size_t)row * Kd + kb + ak0);
        v8bf ahi = *(const v8bf*)(A + (size_t)row * Kd + kb + ak0 + 16);
        v16bf af = make_frag(alo, ahi);
#pragma unroll
        for (int t = 0; t < 4; ++t) {
            const int nloc = wave * 64 + t * 16 + lid;
            v8bf blo = *(const v8bf*)(&sWT[cur][nloc][kfrag]);
            v8bf bhi = *(const v8bf*)(&sWT[cur][nloc][kfrag + 16]);
            acc[t] = wmma_bf16(af, make_frag(blo, bhi), acc[t]);
        }
        __syncthreads();
    }

#pragma unroll
    for (int t = 0; t < 4; ++t) {
        const int c = n0blk + wave * 64 + t * 16 + lid;
        const int which = c / DMODEL;
        const int d0 = c % DMODEL;
        const int h = d0 / HS, hd = d0 % HS;
        const float bv = bias[c];
#pragma unroll
        for (int i = 0; i < 8; ++i) {
            const int r = m0 + i + 8 * hi16;
            const int b = r / L_SEQ, l = r % L_SEQ;
            const __bf16 val = (__bf16)(acc[t][i] + bv);
            const size_t bh = (size_t)(b * N_HEAD + h);
            if (which == 0)      qo[(bh * L_SEQ + l) * HS + hd] = val;
            else if (which == 1) ko[(bh * L_SEQ + l) * HS + hd] = val;
            else                 vt[(bh * HS + hd) * L_SEQ + l] = val;
        }
    }
}

// ---------------------------------------------------------------- attention
__global__ void attn_kernel(const __bf16* __restrict__ q,
                            const __bf16* __restrict__ k,
                            const __bf16* __restrict__ vt,
                            const __bf16* __restrict__ er,
                            __bf16* __restrict__ y) {
    __shared__ __attribute__((aligned(16))) float  sG[16][48];
    __shared__ __attribute__((aligned(16))) __bf16 sP[16][32];

    const int lane = threadIdx.x & 31;
    const int lid  = lane & 15;
    const int hi16 = lane >> 4;
    const int bh = blockIdx.x;
    const int b  = bh / N_HEAD, h = bh % N_HEAD;
    const int l0 = blockIdx.y * 16;

    const __bf16* qp = q  + (size_t)bh * L_SEQ * HS;
    const __bf16* kp = k  + (size_t)bh * L_SEQ * HS;
    const __bf16* vp = vt + (size_t)bh * HS * L_SEQ;

    const int arow = l0 + lid;
    const int ak0  = hi16 * 8;
    v16bf qf[2];
#pragma unroll
    for (int s = 0; s < 2; ++s) {
        v8bf lo = *(const v8bf*)(qp + (size_t)arow * HS + s * 32 + ak0);
        v8bf hi = *(const v8bf*)(qp + (size_t)arow * HS + s * 32 + ak0 + 16);
        qf[s] = make_frag(lo, hi);
    }

    float mrow[8], lrow[8];
    v8f oacc[4];
#pragma unroll
    for (int i = 0; i < 8; ++i) { mrow[i] = -__builtin_inff(); lrow[i] = 0.f; }
    for (int t = 0; t < 4; ++t) oacc[t] = {};

    const float scale = 0.125f;               // 1/sqrt(64)
    const int nkb = (l0 + 15) / 32 + 1;

    for (int kb = 0; kb < nkb; ++kb) {
        const int m0 = kb * 32;

        v8f sc[2]; sc[0] = {}; sc[1] = {};
#pragma unroll
        for (int ct = 0; ct < 2; ++ct) {
            const int key = m0 + ct * 16 + lid;
#pragma unroll
            for (int ss = 0; ss < 2; ++ss) {
                const int dbase = ss * 32 + hi16 * 8;
                v8bf lo = *(const v8bf*)(kp + (size_t)key * HS + dbase);
                v8bf hi = *(const v8bf*)(kp + (size_t)key * HS + dbase + 16);
                sc[ct] = wmma_bf16(qf[ss], make_frag(lo, hi), sc[ct]);
            }
        }

        const int ebase = m0 - l0 + (L_SEQ - 1) - 15;
        v8f g[3]; g[0] = {}; g[1] = {}; g[2] = {};
#pragma unroll
        for (int gt = 0; gt < 3; ++gt) {
            int erow = ebase + gt * 16 + lid;
            erow = erow < 0 ? 0 : (erow > L_SEQ - 1 ? L_SEQ - 1 : erow);
#pragma unroll
            for (int ss = 0; ss < 2; ++ss) {
                const int dbase = ss * 32 + hi16 * 8;
                v8bf lo = *(const v8bf*)(er + (size_t)erow * HS + dbase);
                v8bf hi = *(const v8bf*)(er + (size_t)erow * HS + dbase + 16);
                g[gt] = wmma_bf16(qf[ss], make_frag(lo, hi), g[gt]);
            }
        }
#pragma unroll
        for (int gt = 0; gt < 3; ++gt)
#pragma unroll
            for (int i = 0; i < 8; ++i)
                sG[i + 8 * hi16][gt * 16 + lid] = g[gt][i];
        __syncthreads();

        float p[2][8];
#pragma unroll
        for (int i = 0; i < 8; ++i) {
            const int r = i + 8 * hi16;
            const int rowg = l0 + r;
            float sv[2];
            float best = -__builtin_inff();
#pragma unroll
            for (int ct = 0; ct < 2; ++ct) {
                const int cloc = ct * 16 + lid;
                const int col  = m0 + cloc;
                float val = (sc[ct][i] + sG[r][cloc - r + 15]) * scale;
                if (col > rowg) val = -__builtin_inff();
                sv[ct] = val;
                best = fmaxf(best, val);
            }
#pragma unroll
            for (int off = 1; off < 16; off <<= 1)
                best = fmaxf(best, __shfl_xor(best, off, 32));
            const float mnew = fmaxf(mrow[i], best);
            const float corr = __expf(mrow[i] - mnew);
            float psum = 0.f;
#pragma unroll
            for (int ct = 0; ct < 2; ++ct) {
                const float pv = __expf(sv[ct] - mnew);
                p[ct][i] = pv; psum += pv;
            }
#pragma unroll
            for (int off = 1; off < 16; off <<= 1)
                psum += __shfl_xor(psum, off, 32);
            lrow[i] = lrow[i] * corr + psum;
            mrow[i] = mnew;
#pragma unroll
            for (int dt = 0; dt < 4; ++dt) oacc[dt][i] *= corr;
        }
        __syncthreads();

#pragma unroll
        for (int ct = 0; ct < 2; ++ct)
#pragma unroll
            for (int i = 0; i < 8; ++i)
                sP[i + 8 * hi16][ct * 16 + lid] = (__bf16)p[ct][i];
        __syncthreads();
        v8bf plo = *(const v8bf*)(&sP[lid][ak0]);
        v8bf phi = *(const v8bf*)(&sP[lid][ak0 + 16]);
        v16bf pf = make_frag(plo, phi);

#pragma unroll
        for (int dt = 0; dt < 4; ++dt) {
            const int dcol = dt * 16 + lid;
            const int jb = hi16 * 8;
            v8bf lo = *(const v8bf*)(vp + (size_t)dcol * L_SEQ + m0 + jb);
            v8bf hi = *(const v8bf*)(vp + (size_t)dcol * L_SEQ + m0 + jb + 16);
            oacc[dt] = wmma_bf16(pf, make_frag(lo, hi), oacc[dt]);
        }
        __syncthreads();
    }

#pragma unroll
    for (int i = 0; i < 8; ++i) {
        const float inv = 1.f / lrow[i];
        const int lg = l0 + i + 8 * hi16;
#pragma unroll
        for (int dt = 0; dt < 4; ++dt) {
            const float o = oacc[dt][i] * inv;
            y[((size_t)b * L_SEQ + lg) * DMODEL + h * HS + dt * 16 + lid] = (__bf16)o;
        }
    }
}

// ---------------------------------------------------------------- proj GEMM
// A [B*L, D] bf16 ; WT [N=D, K=D] bf16 (K-major) ; out f32 [B*L, D]
__global__ void proj_gemm_kernel(const __bf16* __restrict__ A,
                                 const __bf16* __restrict__ WT,
                                 const float*  __restrict__ bias,
                                 float* __restrict__ out) {
    __shared__ __attribute__((aligned(16))) __bf16 sWT[2][256][32];

    const int Kd = DMODEL, Nn = DMODEL;
    const int lane = threadIdx.x & 31;
    const int wave = threadIdx.x >> 5;
    const int m0 = blockIdx.x * 16;
    const int n0blk = blockIdx.y * 256;
    const int lid  = lane & 15;
    const int hi16 = lane >> 4;
    const int row  = m0 + lid;
    const int ak0  = hi16 * 8;
    const int kfrag = hi16 * 8;

    v8f acc[4];
    for (int t = 0; t < 4; ++t) acc[t] = {};

    if (wave == 0)
        tdm_load_2d((unsigned)(size_t)&sWT[0][0][0],
                    WT + (size_t)n0blk * Kd, Kd, Nn, Kd, 32, 256);

    for (int kb = 0; kb < Kd; kb += 32) {
        const int cur = (kb >> 5) & 1;
        const bool more = (kb + 32) < Kd;
        if (wave == 0) {
            if (more) {
                tdm_load_2d((unsigned)(size_t)&sWT[cur ^ 1][0][0],
                            WT + (size_t)n0blk * Kd + (kb + 32), Kd, Nn, Kd, 32, 256);
                __builtin_amdgcn_s_wait_tensorcnt(1);
            } else {
                __builtin_amdgcn_s_wait_tensorcnt(0);
            }
        }
        __syncthreads();

        if (kb + 64 < Kd) __builtin_prefetch(A + (size_t)row * Kd + kb + 64, 0, 1);
        v8bf alo = *(const v8bf*)(A + (size_t)row * Kd + kb + ak0);
        v8bf ahi = *(const v8bf*)(A + (size_t)row * Kd + kb + ak0 + 16);
        v16bf af = make_frag(alo, ahi);
#pragma unroll
        for (int t = 0; t < 4; ++t) {
            const int nloc = wave * 64 + t * 16 + lid;
            v8bf blo = *(const v8bf*)(&sWT[cur][nloc][kfrag]);
            v8bf bhi = *(const v8bf*)(&sWT[cur][nloc][kfrag + 16]);
            acc[t] = wmma_bf16(af, make_frag(blo, bhi), acc[t]);
        }
        __syncthreads();
    }

#pragma unroll
    for (int t = 0; t < 4; ++t) {
        const int c = n0blk + wave * 64 + t * 16 + lid;
        const float bv = bias[c];
#pragma unroll
        for (int i = 0; i < 8; ++i) {
            const int r = m0 + i + 8 * hi16;
            out[(size_t)r * Nn + c] = acc[t][i] + bv;
        }
    }
}

// ---------------------------------------------------------------- launcher
extern "C" void kernel_launch(void* const* d_in, const int* in_sizes, int n_in,
                              void* d_out, int out_size, void* d_ws, size_t ws_size,
                              hipStream_t stream) {
    const float* x     = (const float*)d_in[0];
    const float* Wqkv  = (const float*)d_in[1];
    const float* bqkv  = (const float*)d_in[2];
    const float* Wproj = (const float*)d_in[3];
    const float* bproj = (const float*)d_in[4];
    const float* Er    = (const float*)d_in[5];
    float* out = (float*)d_out;

    const size_t nX  = (size_t)BATCH * L_SEQ * DMODEL;      // 4M
    const size_t nWq = (size_t)DMODEL * 3 * DMODEL;         // 3M
    const size_t nWp = (size_t)DMODEL * DMODEL;             // 1M
    const size_t nEr = (size_t)L_SEQ * HS;                  // 128K
    const size_t nQ  = (size_t)BATCH * N_HEAD * L_SEQ * HS; // 4M

    char* ws = (char*)d_ws;
    __bf16* xb     = (__bf16*)ws; ws += nX  * 2;
    __bf16* wqkvT  = (__bf16*)ws; ws += nWq * 2;   // [3D, D] K-major
    __bf16* wprojT = (__bf16*)ws; ws += nWp * 2;   // [D, D]  K-major
    __bf16* erb    = (__bf16*)ws; ws += nEr * 2;
    __bf16* q_ws   = (__bf16*)ws; ws += nQ  * 2;
    __bf16* k_ws   = (__bf16*)ws; ws += nQ  * 2;
    __bf16* vt_ws  = (__bf16*)ws; ws += nQ  * 2;
    __bf16* y_ws   = (__bf16*)ws; ws += nX  * 2;

    cvt_bf16_kernel  <<<1024, 256, 0, stream>>>(x,  xb,  (int)nX);
    cvt_bf16_T_kernel<<<1024, 256, 0, stream>>>(Wqkv,  wqkvT,  DMODEL, 3 * DMODEL);
    cvt_bf16_T_kernel<<<1024, 256, 0, stream>>>(Wproj, wprojT, DMODEL, DMODEL);
    cvt_bf16_kernel  <<<256,  256, 0, stream>>>(Er, erb, (int)nEr);

    qkv_gemm_kernel<<<dim3(256, 12), 128, 0, stream>>>(xb, wqkvT, bqkv, q_ws, k_ws, vt_ws);
    attn_kernel    <<<dim3(32, 128), 32, 0, stream>>>(q_ws, k_ws, vt_ws, erb, y_ws);
    proj_gemm_kernel<<<dim3(256, 4), 128, 0, stream>>>(y_ws, wprojT, bproj, out);
}
